// LocalEmbeddingModule_89189290868836
// MI455X (gfx1250) — compile-verified, matrix operand order checked by reading, not run on hardware
//
// MI455X (gfx1250) implementation of LocalEmbeddingModule forward.
//
// - All dense matmuls -> v_wmma_f32_16x16x32_f16 (f16 operands, f32 accum).
// - Weight tiles staged into LDS by the Tensor Data Mover (tensor_load_to_lds,
//   TENSORcnt-tracked) from pre-transposed f16 weights; fragments then read as
//   contiguous 16B ds_load_b128 pairs matching the CDNA5 16-bit layout.
// - Laplacian propagations as edge-parallel global_atomic_add_f32 scatter.
// - MMFormer/fuse branch is row-wise => evaluated only at the B batch rows.

#include <hip/hip_runtime.h>
#include <hip/hip_bf16.h>
#include <stdint.h>

typedef _Float16 v16h __attribute__((ext_vector_type(16)));
typedef _Float16 v8h  __attribute__((ext_vector_type(8)));
typedef float    v8f  __attribute__((ext_vector_type(8)));
typedef unsigned int u32x4 __attribute__((ext_vector_type(4)));
typedef int          i32x4 __attribute__((ext_vector_type(4)));
typedef int          i32x8 __attribute__((ext_vector_type(8)));

#define D 64

// ---------------------------------------------------------------- utilities
__global__ void zero_f32(float* __restrict__ p, long long n) {
  long long t = (long long)blockIdx.x * blockDim.x + threadIdx.x;
  if (t < n) p[t] = 0.0f;
}

__global__ void f32_to_f16(const float* __restrict__ a, _Float16* __restrict__ b, long long n) {
  long long t = (long long)blockIdx.x * blockDim.x + threadIdx.x;
  if (t < n) b[t] = (_Float16)a[t];
}

__global__ void copy_f32(float* __restrict__ d, const float* __restrict__ s, long long n) {
  long long t = (long long)blockIdx.x * blockDim.x + threadIdx.x;
  if (t < n) d[t] = s[t];
}

__global__ void add_f32(float* __restrict__ d, const float* __restrict__ s, long long n) {
  long long t = (long long)blockIdx.x * blockDim.x + threadIdx.x;
  if (t < n) d[t] += s[t];
}

// WT[n*K + k] = (f16) W[k*Nd + n]   (coalesced read over n)
__global__ void transpose_to_f16(const float* __restrict__ W, _Float16* __restrict__ WT,
                                 int K, int Nd) {
  long long t = (long long)blockIdx.x * blockDim.x + threadIdx.x;
  if (t >= (long long)K * Nd) return;
  int n = (int)(t % Nd), k = (int)(t / Nd);
  WT[(size_t)n * K + k] = (_Float16)W[(size_t)k * Nd + n];
}

// ---------------------------------------------------------------- graph prep
__global__ void count_rows(const int* __restrict__ rows, float* __restrict__ cnt, long long E) {
  long long t = (long long)blockIdx.x * blockDim.x + threadIdx.x;
  if (t < E) atomicAdd(&cnt[rows[t]], 1.0f);
}

__global__ void rsqrt_counts(const float* __restrict__ cnt, float* __restrict__ r, int n) {
  int t = blockIdx.x * blockDim.x + threadIdx.x;
  if (t < n) r[t] = rsqrtf(cnt[t] + 1e-7f);
}

__global__ void edge_vals(const int* __restrict__ rows, const int* __restrict__ cols,
                          const float* __restrict__ r, float* __restrict__ vals, long long E) {
  long long t = (long long)blockIdx.x * blockDim.x + threadIdx.x;
  if (t < E) vals[t] = r[rows[t]] * r[cols[t]];
}

// out[rows[e], :] += vals[e] * h[cols[e], :]  (64 lanes per edge, coalesced gather)
__global__ void spmm_scatter(const int* __restrict__ rows, const int* __restrict__ cols,
                             const float* __restrict__ vals, const float* __restrict__ h,
                             float* __restrict__ out, long long E) {
  long long t = (long long)blockIdx.x * blockDim.x + threadIdx.x;
  long long e = t >> 6;
  int d = (int)(t & 63);
  if (e >= E) return;
  int rr = rows[e], cc = cols[e];
  atomicAdd(&out[(size_t)rr * D + d], vals[e] * h[(size_t)cc * D + d]);
}

// idg = h1 + h2 + id_emb   (in place into h1)
__global__ void combine_idg(float* __restrict__ h1, const float* __restrict__ h2,
                            const float* __restrict__ e0, long long n) {
  long long t = (long long)blockIdx.x * blockDim.x + threadIdx.x;
  if (t < n) h1[t] = h1[t] + h2[t] + e0[t];
}

// ---------------------------------------------------------------- TDM staging
#if defined(__gfx1250__)
// Build a 2D Tensor-DMA descriptor (CDNA5 ISA) and issue
// tensor_load_to_lds: tile = ntile_rows rows of kdim contiguous f16 elements,
// row stride = kdim elements, destination LDS byte address lds_byte_addr.
__device__ __forceinline__ void tdm_load_tile_to_lds(const _Float16* gptr,
                                                     unsigned lds_byte_addr,
                                                     int kdim, int ntile_rows,
                                                     int ndim_total) {
  unsigned long long ga = (unsigned long long)(uintptr_t)gptr;
  u32x4 g0;
  g0[0] = 1u;                                              // count=1 (valid user D#)
  g0[1] = lds_byte_addr;                                   // lds_addr [63:32]
  g0[2] = (unsigned)(ga & 0xffffffffull);                  // global_addr [95:64]
  g0[3] = (unsigned)((ga >> 32) & 0x1ffffffull) | (2u << 30); // addr[56:32] | type=2
  unsigned t0 = (unsigned)kdim;        // tensor_dim0 (elements)
  unsigned t1 = (unsigned)ndim_total;  // tensor_dim1
  i32x8 g1;
  g1[0] = (int)(1u << 16);                                 // data_size=1 -> 2 bytes
  g1[1] = (int)((t0 & 0xffffu) << 16);                     // tensor_dim0[15:0] @ bits48
  g1[2] = (int)(((t0 >> 16) & 0xffffu) | ((t1 & 0xffffu) << 16));
  g1[3] = (int)(((t1 >> 16) & 0xffffu) | (((unsigned)kdim & 0xffffu) << 16)); // tile_dim0
  g1[4] = (int)((unsigned)ntile_rows & 0xffffu);           // tile_dim1 (tile_dim2=0)
  g1[5] = (int)(unsigned)kdim;                             // tensor_dim0_stride[31:0]
  g1[6] = 0;                                               // stride hi / dim1_stride lo
  g1[7] = 0;
  i32x4 gz = {0, 0, 0, 0};                                 // 2D tensor: groups 2/3 unused
#if __clang_major__ >= 23
  i32x8 gz8 = {0, 0, 0, 0, 0, 0, 0, 0};
  __builtin_amdgcn_tensor_load_to_lds(g0, g1, gz, gz, gz8, 0);
#else
  __builtin_amdgcn_tensor_load_to_lds(g0, g1, gz, gz, 0);
#endif
}
#endif

// ---------------------------------------------------------------- WMMA GEMM
// C[M,Ndim] = act(A[M,KDIM] @ W + bias). A f16; WT = f16 weights pre-transposed
// to [Ndim][KDIM] so the TDM drops the 64-column strip into LDS in exactly the
// [n][k] layout the B fragments want (two contiguous 16B runs per chunk).
// Block: 256 threads = 8 waves; wave -> 16x64 output tile (4 WMMA tiles).
template <int KDIM, bool RELU>
__global__ __launch_bounds__(256) void gemm_wmma_kernel(
    const _Float16* __restrict__ A, const _Float16* __restrict__ WT,
    const float* __restrict__ bias, float* __restrict__ Cf,
    _Float16* __restrict__ Ch, int M, int Ndim) {
  __shared__ _Float16 Wl[64 * KDIM] __attribute__((aligned(16)));
  __shared__ float bl[64];
  const int tid = threadIdx.x;
  const int ncol0 = blockIdx.y * 64;

#if defined(__gfx1250__)
  if (tid < 32) {   // one wave issues the TDM transfer, waits on TENSORcnt
    tdm_load_tile_to_lds(WT + (size_t)ncol0 * KDIM, (unsigned)(uintptr_t)Wl,
                         KDIM, 64, Ndim);
    __builtin_amdgcn_s_wait_tensorcnt(0);
  }
#else
  for (int idx = tid; idx < 64 * KDIM; idx += 256)
    Wl[idx] = WT[(size_t)ncol0 * KDIM + idx];
#endif
  if (tid < 64) bl[tid] = bias[ncol0 + tid];
  __syncthreads();

  const int wave = tid >> 5, lane = tid & 31;
  const int row0 = blockIdx.x * 128 + wave * 16;
  const int m = lane & 15, kb = (lane >> 4) * 8, cl = lane & 15;
  const int row = row0 + m;
  const _Float16* arow = A + (size_t)row * KDIM;
  if (row < M) __builtin_prefetch(arow, 0, 3);            // global_prefetch_b8

  v8f acc[4] = {v8f{}, v8f{}, v8f{}, v8f{}};

  for (int kc = 0; kc < KDIM; kc += 32) {
    v16h af = {};
    if (row < M) {
      v8h alo = *(const v8h*)(arow + kc + kb);
      v8h ahi = *(const v8h*)(arow + kc + 16 + kb);
#pragma unroll
      for (int j = 0; j < 8; ++j) { af[j] = alo[j]; af[8 + j] = ahi[j]; }
    }
#pragma unroll
    for (int nt = 0; nt < 4; ++nt) {
      const _Float16* wp = Wl + (size_t)(nt * 16 + cl) * KDIM + kc + kb;
      v8h blo = *(const v8h*)(wp);
      v8h bhi = *(const v8h*)(wp + 16);
      v16h bf;
#pragma unroll
      for (int j = 0; j < 8; ++j) { bf[j] = blo[j]; bf[8 + j] = bhi[j]; }
      acc[nt] = __builtin_amdgcn_wmma_f32_16x16x32_f16(
          false, af, false, bf, (short)0, acc[nt], false, false);
    }
  }

  const int rbase = row0 + (lane >> 4) * 8;               // C layout: M = 8*(lane>>4)+i
#pragma unroll
  for (int nt = 0; nt < 4; ++nt) {
    const int col = ncol0 + nt * 16 + cl;
    const float bv = bl[nt * 16 + cl];
#pragma unroll
    for (int i = 0; i < 8; ++i) {
      int rr = rbase + i;
      if (rr < M) {
        float val = acc[nt][i] + bv;
        if (RELU) val = fmaxf(val, 0.0f);
        if (Cf) Cf[(size_t)rr * Ndim + col] = val;
        if (Ch) Ch[(size_t)rr * Ndim + col] = (_Float16)val;
      }
    }
  }
}

// ---------------------------------------------------------------- batch MMFormer helpers
__global__ void build_tok(const float* __restrict__ queries, const float* __restrict__ vfeat,
                          const float* __restrict__ tfeat, const int* __restrict__ ids,
                          _Float16* __restrict__ tok, int Bn) {
  long long t = (long long)blockIdx.x * blockDim.x + threadIdx.x;
  if (t >= (long long)Bn * 4 * D) return;
  int d = (int)(t & 63);
  int r = (int)(t >> 6);
  int bi = r >> 2, tk = r & 3;
  float v;
  if (tk < 2)       v = queries[tk * D + d];
  else if (tk == 2) v = vfeat[(size_t)ids[bi] * D + d];
  else              v = tfeat[(size_t)ids[bi] * D + d];
  tok[t] = (_Float16)v;
}

// one thread per (node, head): 4x4 attention over T=4 tokens, dh=16
__global__ void attn_kernel(const _Float16* __restrict__ q, const _Float16* __restrict__ k,
                            const _Float16* __restrict__ v, _Float16* __restrict__ o, int Bn) {
  int t = blockIdx.x * blockDim.x + threadIdx.x;
  if (t >= Bn * 4) return;
  int bi = t >> 2, h = t & 3;
  size_t base = (size_t)bi * 4 * D + h * 16;
  float S[4][4];
#pragma unroll
  for (int ti = 0; ti < 4; ++ti)
#pragma unroll
    for (int si = 0; si < 4; ++si) {
      float acc = 0.f;
      for (int d = 0; d < 16; ++d)
        acc += (float)q[base + ti * D + d] * (float)k[base + si * D + d];
      S[ti][si] = acc * 0.25f;  // 1/sqrt(16)
    }
#pragma unroll
  for (int ti = 0; ti < 4; ++ti) {
    float mx = S[ti][0];
    for (int si = 1; si < 4; ++si) mx = fmaxf(mx, S[ti][si]);
    float e[4], sum = 0.f;
    for (int si = 0; si < 4; ++si) { e[si] = __expf(S[ti][si] - mx); sum += e[si]; }
    float inv = 1.0f / sum;
    for (int d = 0; d < 16; ++d) {
      float ov = 0.f;
      for (int si = 0; si < 4; ++si) ov += e[si] * inv * (float)v[base + si * D + d];
      o[base + ti * D + d] = (_Float16)ov;
    }
  }
}

__global__ void ln1_kernel(const float* __restrict__ attn, const float* __restrict__ queries,
                           const float* __restrict__ vfeat, const float* __restrict__ tfeat,
                           const int* __restrict__ ids, const float* __restrict__ g,
                           const float* __restrict__ bb, float* __restrict__ xf,
                           _Float16* __restrict__ xh, int Bn) {
  int r = blockIdx.x * blockDim.x + threadIdx.x;
  if (r >= Bn * 4) return;
  int bi = r >> 2, tk = r & 3;
  const float* src = (tk < 2) ? (queries + tk * D)
                   : (tk == 2 ? vfeat + (size_t)ids[bi] * D : tfeat + (size_t)ids[bi] * D);
  float v[D], s = 0.f;
  for (int d = 0; d < D; ++d) { v[d] = src[d] + attn[(size_t)r * D + d]; s += v[d]; }
  float mean = s * (1.0f / D), var = 0.f;
  for (int d = 0; d < D; ++d) { float c = v[d] - mean; var += c * c; }
  float inv = rsqrtf(var * (1.0f / D) + 1e-5f);
  for (int d = 0; d < D; ++d) {
    float y = (v[d] - mean) * inv * g[d] + bb[d];
    xf[(size_t)r * D + d] = y;
    xh[(size_t)r * D + d] = (_Float16)y;
  }
}

// x2 = LN(x + ffn_out), tokens t<2 only, packed [b, t*64+d] as f16
__global__ void ln2_kernel(const float* __restrict__ x, const float* __restrict__ y,
                           const float* __restrict__ g, const float* __restrict__ bb,
                           _Float16* __restrict__ fuse_in, int Bn) {
  int idx = blockIdx.x * blockDim.x + threadIdx.x;
  if (idx >= Bn * 2) return;
  int bi = idx >> 1, tk = idx & 1;
  size_t r = (size_t)bi * 4 + tk;
  float v[D], s = 0.f;
  for (int d = 0; d < D; ++d) { v[d] = x[r * D + d] + y[r * D + d]; s += v[d]; }
  float mean = s * (1.0f / D), var = 0.f;
  for (int d = 0; d < D; ++d) { float c = v[d] - mean; var += c * c; }
  float inv = rsqrtf(var * (1.0f / D) + 1e-5f);
  for (int d = 0; d < D; ++d)
    fuse_in[(size_t)bi * 128 + tk * D + d] = (_Float16)((v[d] - mean) * inv * g[d] + bb[d]);
}

__global__ void l2norm_rows(float* __restrict__ x, int rows) {
  int r = blockIdx.x * blockDim.x + threadIdx.x;
  if (r >= rows) return;
  float s = 0.f;
  for (int d = 0; d < D; ++d) { float v = x[(size_t)r * D + d]; s += v * v; }
  float inv = 1.0f / fmaxf(sqrtf(s), 1e-12f);
  for (int d = 0; d < D; ++d) x[(size_t)r * D + d] *= inv;
}

__global__ void gather_rows_f16(const float* __restrict__ src, const int* __restrict__ ids,
                                _Float16* __restrict__ dst, int Bn) {
  long long t = (long long)blockIdx.x * blockDim.x + threadIdx.x;
  if (t >= (long long)Bn * D) return;
  int bi = (int)(t >> 6), d = (int)(t & 63);
  dst[t] = (_Float16)src[(size_t)ids[bi] * D + d];
}

__global__ void final_out(const float* __restrict__ idlin, const float* __restrict__ fuse,
                          const float* __restrict__ vfeat, const float* __restrict__ tfeat,
                          const int* __restrict__ ids, float* __restrict__ out0, int Bn) {
  long long t = (long long)blockIdx.x * blockDim.x + threadIdx.x;
  if (t >= (long long)Bn * D) return;
  int bi = (int)(t >> 6), d = (int)(t & 63);
  size_t g = (size_t)ids[bi] * D + d;
  out0[t] = idlin[t] + vfeat[g] + tfeat[g] + fuse[t];
}

// ---------------------------------------------------------------- launcher
static inline unsigned nblk(long long n, int b) { return (unsigned)((n + b - 1) / b); }

extern "C" void kernel_launch(void* const* d_in, const int* in_sizes, int n_in,
                              void* d_out, int out_size, void* d_ws, size_t ws_size,
                              hipStream_t stream) {
  (void)n_in; (void)out_size; (void)ws_size;
  const int*   item_ids = (const int*)d_in[0];
  const float* id_emb   = (const float*)d_in[1];
  const float* v_feat   = (const float*)d_in[2];
  const float* t_feat   = (const float*)d_in[3];
  const int*   id_rows  = (const int*)d_in[4];
  const int*   id_cols  = (const int*)d_in[5];
  const int*   mm_rows  = (const int*)d_in[6];
  const int*   mm_cols  = (const int*)d_in[7];
  const float* Wv0 = (const float*)d_in[8],  *bv0 = (const float*)d_in[9];
  const float* Wv1 = (const float*)d_in[10], *bv1 = (const float*)d_in[11];
  const float* Wt0 = (const float*)d_in[12], *bt0 = (const float*)d_in[13];
  const float* Wt1 = (const float*)d_in[14], *bt1 = (const float*)d_in[15];
  const float* Wid = (const float*)d_in[16], *bid = (const float*)d_in[17];
  const float* Wmm = (const float*)d_in[18], *bmm = (const float*)d_in[19];
  const float* queries = (const float*)d_in[20];
  const float* wq = (const float*)d_in[21], *bq = (const float*)d_in[22];
  const float* wk = (const float*)d_in[23], *bk = (const float*)d_in[24];
  const float* wv = (const float*)d_in[25], *bvv = (const float*)d_in[26];
  const float* wo = (const float*)d_in[27], *bo = (const float*)d_in[28];
  const float* ln1g = (const float*)d_in[29], *ln1b = (const float*)d_in[30];
  const float* ln2g = (const float*)d_in[31], *ln2b = (const float*)d_in[32];
  const float* w1 = (const float*)d_in[33], *b1 = (const float*)d_in[34];
  const float* w2 = (const float*)d_in[35], *b2 = (const float*)d_in[36];

  const int B = in_sizes[0];
  const long long ND = in_sizes[1];        // N*64
  const int N = (int)(ND / D);
  const long long E_id = in_sizes[4];
  const long long E_mm = in_sizes[6];

  float* out0  = (float*)d_out;            // [B,64]
  float* v_out = out0 + (size_t)B * D;     // [N,64]
  float* t_out = v_out + (size_t)N * D;    // [N,64]

  // ---- workspace carve (256B aligned)
  char* ws = (char*)d_ws; size_t off = 0;
  auto carve = [&](size_t bytes) -> char* {
    char* p = ws + off; off = (off + bytes + 255) & ~(size_t)255; return p;
  };
  float*    cnt      = (float*)carve((size_t)N * 4);
  float*    rnorm    = (float*)carve((size_t)N * 4);
  float*    id_vals  = (float*)carve((size_t)E_id * 4);
  float*    mm_vals  = (float*)carve((size_t)E_mm * 4);
  float*    bufA     = (float*)carve((size_t)ND * 4);
  float*    bufB     = (float*)carve((size_t)ND * 4);
  _Float16* buf16    = (_Float16*)carve((size_t)ND * 2);
  const long long M4 = (long long)B * 4;   // token rows
  _Float16* tok16    = (_Float16*)carve((size_t)M4 * D * 2);
  _Float16* q16      = (_Float16*)carve((size_t)M4 * D * 2);
  _Float16* k16      = (_Float16*)carve((size_t)M4 * D * 2);
  _Float16* v16      = (_Float16*)carve((size_t)M4 * D * 2);
  _Float16* o16      = (_Float16*)carve((size_t)M4 * D * 2);
  float*    tmpM     = (float*)carve((size_t)M4 * D * 4);    // attn-out, then ffn-out
  float*    xB       = (float*)carve((size_t)M4 * D * 4);
  _Float16* xB16     = (_Float16*)carve((size_t)M4 * D * 2);
  _Float16* hid16    = (_Float16*)carve((size_t)M4 * 256 * 2);
  _Float16* fusein16 = (_Float16*)carve((size_t)B * 128 * 2);
  float*    fuseF    = (float*)carve((size_t)B * D * 4);
  _Float16* idg16    = (_Float16*)carve((size_t)B * D * 2);
  float*    idlin    = (float*)carve((size_t)B * D * 4);
  // pre-transposed f16 weights [Ndim][KDIM] for TDM staging
  _Float16* Tv0 = (_Float16*)carve(4096 * 2);
  _Float16* Tv1 = (_Float16*)carve(4096 * 2);
  _Float16* Tt0 = (_Float16*)carve(4096 * 2);
  _Float16* Tt1 = (_Float16*)carve(4096 * 2);
  _Float16* Tid = (_Float16*)carve(4096 * 2);
  _Float16* Tq  = (_Float16*)carve(4096 * 2);
  _Float16* Tk  = (_Float16*)carve(4096 * 2);
  _Float16* Tv  = (_Float16*)carve(4096 * 2);
  _Float16* To  = (_Float16*)carve(4096 * 2);
  _Float16* T1  = (_Float16*)carve(16384 * 2);   // w1: [256][64]
  _Float16* T2  = (_Float16*)carve(16384 * 2);   // w2: [64][256]
  _Float16* Tmm = (_Float16*)carve(8192 * 2);    // mm_proj: [64][128]

  const int TB = 256;
  dim3 blk(TB);
  const unsigned gx128_N  = (unsigned)((N + 127) / 128);
  const unsigned gx128_M4 = (unsigned)((M4 + 127) / 128);

  // ---- Phase 0: pre-transpose all GEMM weights to f16 [Ndim][KDIM]
  transpose_to_f16<<<nblk(4096, TB), blk, 0, stream>>>(Wv0, Tv0, 64, 64);
  transpose_to_f16<<<nblk(4096, TB), blk, 0, stream>>>(Wv1, Tv1, 64, 64);
  transpose_to_f16<<<nblk(4096, TB), blk, 0, stream>>>(Wt0, Tt0, 64, 64);
  transpose_to_f16<<<nblk(4096, TB), blk, 0, stream>>>(Wt1, Tt1, 64, 64);
  transpose_to_f16<<<nblk(4096, TB), blk, 0, stream>>>(Wid, Tid, 64, 64);
  transpose_to_f16<<<nblk(4096, TB), blk, 0, stream>>>(wq, Tq, 64, 64);
  transpose_to_f16<<<nblk(4096, TB), blk, 0, stream>>>(wk, Tk, 64, 64);
  transpose_to_f16<<<nblk(4096, TB), blk, 0, stream>>>(wv, Tv, 64, 64);
  transpose_to_f16<<<nblk(4096, TB), blk, 0, stream>>>(wo, To, 64, 64);
  transpose_to_f16<<<nblk(16384, TB), blk, 0, stream>>>(w1, T1, 64, 256);
  transpose_to_f16<<<nblk(16384, TB), blk, 0, stream>>>(w2, T2, 256, 64);
  transpose_to_f16<<<nblk(8192, TB), blk, 0, stream>>>(Wmm, Tmm, 128, 64);

  // ---- Phase A: normalized-Laplacian edge values for both graphs
  zero_f32<<<nblk(N, TB), blk, 0, stream>>>(cnt, N);
  count_rows<<<nblk(E_id, TB), blk, 0, stream>>>(id_rows, cnt, E_id);
  rsqrt_counts<<<nblk(N, TB), blk, 0, stream>>>(cnt, rnorm, N);
  edge_vals<<<nblk(E_id, TB), blk, 0, stream>>>(id_rows, id_cols, rnorm, id_vals, E_id);
  zero_f32<<<nblk(N, TB), blk, 0, stream>>>(cnt, N);
  count_rows<<<nblk(E_mm, TB), blk, 0, stream>>>(mm_rows, cnt, E_mm);
  rsqrt_counts<<<nblk(N, TB), blk, 0, stream>>>(cnt, rnorm, N);
  edge_vals<<<nblk(E_mm, TB), blk, 0, stream>>>(mm_rows, mm_cols, rnorm, mm_vals, E_mm);

  // ---- Phase B/C: GCN stacks (reference bug kept: mm graph for BOTH modalities)
  struct { const float* feat; const _Float16* W0; const float* b0;
           const _Float16* W1; const float* b1; float* out; }
    mods[2] = { { v_feat, Tv0, bv0, Tv1, bv1, v_out },
                { t_feat, Tt0, bt0, Tt1, bt1, t_out } };
  for (int mmod = 0; mmod < 2; ++mmod) {
    zero_f32<<<nblk(ND, TB), blk, 0, stream>>>(bufA, ND);
    spmm_scatter<<<nblk(E_mm * D, TB), blk, 0, stream>>>(mm_rows, mm_cols, mm_vals,
                                                         mods[mmod].feat, bufA, E_mm);
    f32_to_f16<<<nblk(ND, TB), blk, 0, stream>>>(bufA, buf16, ND);
    gemm_wmma_kernel<64, true><<<dim3(gx128_N, 1), blk, 0, stream>>>(
        buf16, mods[mmod].W0, mods[mmod].b0, bufB, (_Float16*)nullptr, N, D);
    copy_f32<<<nblk(ND, TB), blk, 0, stream>>>(mods[mmod].out, bufB, ND);
    zero_f32<<<nblk(ND, TB), blk, 0, stream>>>(bufA, ND);
    spmm_scatter<<<nblk(E_mm * D, TB), blk, 0, stream>>>(mm_rows, mm_cols, mm_vals,
                                                         bufB, bufA, E_mm);
    f32_to_f16<<<nblk(ND, TB), blk, 0, stream>>>(bufA, buf16, ND);
    gemm_wmma_kernel<64, true><<<dim3(gx128_N, 1), blk, 0, stream>>>(
        buf16, mods[mmod].W1, mods[mmod].b1, bufB, (_Float16*)nullptr, N, D);
    add_f32<<<nblk(ND, TB), blk, 0, stream>>>(mods[mmod].out, bufB, ND);
  }

  // ---- Phase D: buildItemGraph (2-hop id-graph propagation) + id_lin at batch rows
  zero_f32<<<nblk(ND, TB), blk, 0, stream>>>(bufA, ND);
  spmm_scatter<<<nblk(E_id * D, TB), blk, 0, stream>>>(id_rows, id_cols, id_vals,
                                                       id_emb, bufA, E_id);
  zero_f32<<<nblk(ND, TB), blk, 0, stream>>>(bufB, ND);
  spmm_scatter<<<nblk(E_id * D, TB), blk, 0, stream>>>(id_rows, id_cols, id_vals,
                                                       bufA, bufB, E_id);
  combine_idg<<<nblk(ND, TB), blk, 0, stream>>>(bufA, bufB, id_emb, ND);  // idg in bufA
  gather_rows_f16<<<nblk((long long)B * D, TB), blk, 0, stream>>>(bufA, item_ids, idg16, B);
  gemm_wmma_kernel<64, false><<<dim3(nblk(B, 128), 1), blk, 0, stream>>>(
      idg16, Tid, bid, idlin, (_Float16*)nullptr, B, D);
  l2norm_rows<<<nblk(B, TB), blk, 0, stream>>>(idlin, B);

  // ---- Phase E: MMFormer fusion, batch rows only (row-wise in the reference)
  build_tok<<<nblk(M4 * D, TB), blk, 0, stream>>>(queries, v_feat, t_feat, item_ids, tok16, B);
  gemm_wmma_kernel<64, false><<<dim3(gx128_M4, 1), blk, 0, stream>>>(
      tok16, Tq, bq, (float*)nullptr, q16, (int)M4, D);
  gemm_wmma_kernel<64, false><<<dim3(gx128_M4, 1), blk, 0, stream>>>(
      tok16, Tk, bk, (float*)nullptr, k16, (int)M4, D);
  gemm_wmma_kernel<64, false><<<dim3(gx128_M4, 1), blk, 0, stream>>>(
      tok16, Tv, bvv, (float*)nullptr, v16, (int)M4, D);
  attn_kernel<<<nblk(M4, TB), blk, 0, stream>>>(q16, k16, v16, o16, B);
  gemm_wmma_kernel<64, false><<<dim3(gx128_M4, 1), blk, 0, stream>>>(
      o16, To, bo, tmpM, (_Float16*)nullptr, (int)M4, D);
  ln1_kernel<<<nblk(M4, TB), blk, 0, stream>>>(tmpM, queries, v_feat, t_feat, item_ids,
                                               ln1g, ln1b, xB, xB16, B);
  gemm_wmma_kernel<64, true><<<dim3(gx128_M4, 4), blk, 0, stream>>>(
      xB16, T1, b1, (float*)nullptr, hid16, (int)M4, 256);
  gemm_wmma_kernel<256, false><<<dim3(gx128_M4, 1), blk, 0, stream>>>(
      hid16, T2, b2, tmpM, (_Float16*)nullptr, (int)M4, D);
  ln2_kernel<<<nblk((long long)B * 2, TB), blk, 0, stream>>>(xB, tmpM, ln2g, ln2b, fusein16, B);
  gemm_wmma_kernel<128, false><<<dim3(nblk(B, 128), 1), blk, 0, stream>>>(
      fusein16, Tmm, bmm, fuseF, (_Float16*)nullptr, B, D);
  l2norm_rows<<<nblk(B, TB), blk, 0, stream>>>(fuseF, B);

  // ---- Phase F: fuse_emb[item_ids]
  final_out<<<nblk((long long)B * D, TB), blk, 0, stream>>>(idlin, fuseF, v_feat, t_feat,
                                                            item_ids, out0, B);
}
